// StarTransformer_28664611734048
// MI455X (gfx1250) — matrix operand order repaired and checked by reading, not compiled, not run on hardware
//
#include <hip/hip_runtime.h>
#include <stdint.h>

// ---------------- problem constants ----------------
#define B_      4
#define L_      2048
#define H_      1024
#define D_      1024
#define NH_     16
#define HD_     64
#define ITERS_  2
#define LP_     (L_ + 1)          // 2049
#define ROWS_X  (B_ * L_)         // 8192 positions
#define ROWS_Y  (B_ * LP_)        // 8196 rows (relay + nodes per batch)
#define ROWS_Y_PAD 8320           // 65 * 128 (GEMM M-tile padded)
#define EPS_    1e-6f
#define SLOPE_  0.2f

typedef __attribute__((ext_vector_type(16))) __bf16 v16bf;
typedef __attribute__((ext_vector_type(8)))  float  v8f;

static __device__ __forceinline__ uint16_t f2bf(float f) {
    uint32_t u = __float_as_uint(f);
    uint32_t r = (u + 0x7FFFu + ((u >> 16) & 1u)) >> 16;   // round-to-nearest-even
    return (uint16_t)r;
}
static __device__ __forceinline__ float bf2f(uint16_t h) {
    return __uint_as_float(((uint32_t)h) << 16);
}
static __device__ __forceinline__ float leaky(float v) {
    return v > 0.f ? v : SLOPE_ * v;
}

union FragU { uint4 u[2]; v16bf v; };
union V4U  { uint4 u; uint16_t h[8]; };

// CDNA5 async direct-to-LDS copy (16B per lane), tracked by ASYNCcnt.
static __device__ __forceinline__ void async_b128(uint32_t lds_off, const uint4* g) {
    asm volatile("global_load_async_to_lds_b128 %0, %1, off"
                 :: "v"(lds_off), "v"((unsigned long long)(uintptr_t)g)
                 : "memory");
}
static __device__ __forceinline__ void wait_async0() {
    asm volatile("s_wait_asynccnt 0x0" ::: "memory");
}
static __device__ __forceinline__ uint32_t lds_off_of(const void* p) {
    return (uint32_t)(uintptr_t)p;   // low 32 bits of generic LDS address = LDS offset
}

// ================= WMMA GEMM (single) =================
// C[M x 1024] = A[M x 1024](bf16) * Wt(bf16, [n*1024+k]) + bias
// block tile 128x128, 8 waves each 64x32; BK=32; async double-buffered staging.
template <bool OUT_BF16, bool LEAKY>
__global__ __launch_bounds__(256)
void gemm_bf16(const uint16_t* __restrict__ A,
               const uint16_t* __restrict__ Wt,
               const float*    __restrict__ bias,
               void*           __restrict__ Cout,
               int Mvalid) {
    __shared__ uint4 Alds[2][128 * 5];
    __shared__ uint4 Blds[2][128 * 5];

    const int tid  = threadIdx.x;
    const int lane = tid & 31;
    const int half = lane >> 4;
    const int wid  = tid >> 5;
    const int wm   = wid & 1;     // 2 x 64-row sub-blocks
    const int wn   = wid >> 1;    // 4 x 32-col sub-blocks
    const int rowBase = blockIdx.y * 128;
    const int colBase = blockIdx.x * 128;

    v8f acc[4][2];
#pragma unroll
    for (int i = 0; i < 4; ++i)
#pragma unroll
        for (int j = 0; j < 2; ++j)
#pragma unroll
            for (int r = 0; r < 8; ++r) acc[i][j][r] = 0.f;

    const uint4* Ag = (const uint4*)A;
    const uint4* Bg = (const uint4*)Wt;

    auto stage = [&](int buf, int k0) {
#pragma unroll
        for (int pck = 0; pck < 2; ++pck) {
            int idx = tid + pck * 256;
            int row = idx >> 2, seg = idx & 3;
            async_b128(lds_off_of(&Alds[buf][row * 5 + seg]),
                       Ag + (((size_t)(rowBase + row) << 10) + k0) / 8 + seg);
            async_b128(lds_off_of(&Blds[buf][row * 5 + seg]),
                       Bg + (((size_t)(colBase + row) << 10) + k0) / 8 + seg);
        }
    };

    stage(0, 0);
    for (int s = 0; s < 32; ++s) {
        wait_async0();          // this wave's stage-s copies done
        __syncthreads();        // all waves' stage-s copies done; prior reads retired
        if (s + 1 < 32) stage((s + 1) & 1, (s + 1) * 32);

        const uint4* Ab = Alds[s & 1];
        const uint4* Bb = Blds[s & 1];
        FragU a[4], b[2];
#pragma unroll
        for (int mi = 0; mi < 4; ++mi) {
            int row = wm * 64 + mi * 16 + (lane & 15);
            a[mi].u[0] = Ab[row * 5 + half];
            a[mi].u[1] = Ab[row * 5 + 2 + half];
        }
#pragma unroll
        for (int ni = 0; ni < 2; ++ni) {
            int n = wn * 32 + ni * 16 + (lane & 15);
            b[ni].u[0] = Bb[n * 5 + half];
            b[ni].u[1] = Bb[n * 5 + 2 + half];
        }
#pragma unroll
        for (int mi = 0; mi < 4; ++mi)
#pragma unroll
            for (int ni = 0; ni < 2; ++ni)
                acc[mi][ni] = __builtin_amdgcn_wmma_f32_16x16x32_bf16(
                    false, a[mi].v, false, b[ni].v, (short)0, acc[mi][ni], false, false);
    }

#pragma unroll
    for (int mi = 0; mi < 4; ++mi) {
#pragma unroll
        for (int ni = 0; ni < 2; ++ni) {
            int col = colBase + wn * 32 + ni * 16 + (lane & 15);
            float bv = bias[col];
#pragma unroll
            for (int r = 0; r < 8; ++r) {
                int row = rowBase + wm * 64 + mi * 16 + half * 8 + r;
                if (row < Mvalid) {
                    float v = acc[mi][ni][r] + bv;
                    if constexpr (LEAKY) v = leaky(v);
                    if constexpr (OUT_BF16)
                        ((uint16_t*)Cout)[(size_t)row * 1024 + col] = f2bf(v);
                    else
                        ((float*)Cout)[(size_t)row * 1024 + col] = v;
                }
            }
        }
    }
}

// ================= WMMA GEMM (dual A, shared weight) =================
// C1 = A1*Wt+bias ; C2 = A2*Wt+bias. Block tile 128x64 per A; wave 32x32 per A.
__global__ __launch_bounds__(256)
void gemm_bf16_dualA(const uint16_t* __restrict__ A1, const uint16_t* __restrict__ A2,
                     const uint16_t* __restrict__ Wt, const float* __restrict__ bias,
                     uint16_t* __restrict__ C1, uint16_t* __restrict__ C2) {
    __shared__ uint4 A1lds[2][128 * 5];
    __shared__ uint4 A2lds[2][128 * 5];
    __shared__ uint4 Blds[2][64 * 5];

    const int tid  = threadIdx.x;
    const int lane = tid & 31;
    const int half = lane >> 4;
    const int wid  = tid >> 5;
    const int wm   = wid & 3;     // 4 x 32-row sub-blocks
    const int wn   = wid >> 2;    // 2 x 32-col sub-blocks
    const int rowBase = blockIdx.y * 128;
    const int colBase = blockIdx.x * 64;

    v8f acc1[2][2], acc2[2][2];
#pragma unroll
    for (int i = 0; i < 2; ++i)
#pragma unroll
        for (int j = 0; j < 2; ++j)
#pragma unroll
            for (int r = 0; r < 8; ++r) { acc1[i][j][r] = 0.f; acc2[i][j][r] = 0.f; }

    const uint4* A1g = (const uint4*)A1;
    const uint4* A2g = (const uint4*)A2;
    const uint4* Bg  = (const uint4*)Wt;

    auto stage = [&](int buf, int k0) {
#pragma unroll
        for (int pck = 0; pck < 2; ++pck) {
            int idx = tid + pck * 256;
            int row = idx >> 2, seg = idx & 3;
            size_t goff = (((size_t)(rowBase + row) << 10) + k0) / 8 + seg;
            async_b128(lds_off_of(&A1lds[buf][row * 5 + seg]), A1g + goff);
            async_b128(lds_off_of(&A2lds[buf][row * 5 + seg]), A2g + goff);
        }
        {
            int n = tid >> 2, seg = tid & 3;
            async_b128(lds_off_of(&Blds[buf][n * 5 + seg]),
                       Bg + (((size_t)(colBase + n) << 10) + k0) / 8 + seg);
        }
    };

    stage(0, 0);
    for (int s = 0; s < 32; ++s) {
        wait_async0();
        __syncthreads();
        if (s + 1 < 32) stage((s + 1) & 1, (s + 1) * 32);

        const uint4* A1b = A1lds[s & 1];
        const uint4* A2b = A2lds[s & 1];
        const uint4* Bb  = Blds[s & 1];
        FragU a1[2], a2[2], b[2];
#pragma unroll
        for (int mi = 0; mi < 2; ++mi) {
            int row = wm * 32 + mi * 16 + (lane & 15);
            a1[mi].u[0] = A1b[row * 5 + half];
            a1[mi].u[1] = A1b[row * 5 + 2 + half];
            a2[mi].u[0] = A2b[row * 5 + half];
            a2[mi].u[1] = A2b[row * 5 + 2 + half];
        }
#pragma unroll
        for (int ni = 0; ni < 2; ++ni) {
            int n = wn * 32 + ni * 16 + (lane & 15);
            b[ni].u[0] = Bb[n * 5 + half];
            b[ni].u[1] = Bb[n * 5 + 2 + half];
        }
#pragma unroll
        for (int mi = 0; mi < 2; ++mi)
#pragma unroll
            for (int ni = 0; ni < 2; ++ni) {
                acc1[mi][ni] = __builtin_amdgcn_wmma_f32_16x16x32_bf16(
                    false, a1[mi].v, false, b[ni].v, (short)0, acc1[mi][ni], false, false);
                acc2[mi][ni] = __builtin_amdgcn_wmma_f32_16x16x32_bf16(
                    false, a2[mi].v, false, b[ni].v, (short)0, acc2[mi][ni], false, false);
            }
    }

#pragma unroll
    for (int mi = 0; mi < 2; ++mi) {
#pragma unroll
        for (int ni = 0; ni < 2; ++ni) {
            int col = colBase + wn * 32 + ni * 16 + (lane & 15);
            float bv = bias[col];
#pragma unroll
            for (int r = 0; r < 8; ++r) {
                int row = rowBase + wm * 32 + mi * 16 + half * 8 + r;
                C1[(size_t)row * 1024 + col] = f2bf(acc1[mi][ni][r] + bv);
                C2[(size_t)row * 1024 + col] = f2bf(acc2[mi][ni][r] + bv);
            }
        }
    }
}

// ================= WMMA GEMM (dual B, shared activation) =================
// C1 = A*W1t+b1 ; C2 = A*W2t+b2. Block tile 128x64; wave 32x32 against each B.
__global__ __launch_bounds__(256)
void gemm_bf16_dualB(const uint16_t* __restrict__ A,
                     const uint16_t* __restrict__ W1t, const uint16_t* __restrict__ W2t,
                     const float* __restrict__ b1, const float* __restrict__ b2,
                     uint16_t* __restrict__ C1, uint16_t* __restrict__ C2, int Mvalid) {
    __shared__ uint4 Alds[2][128 * 5];
    __shared__ uint4 B1lds[2][64 * 5];
    __shared__ uint4 B2lds[2][64 * 5];

    const int tid  = threadIdx.x;
    const int lane = tid & 31;
    const int half = lane >> 4;
    const int wid  = tid >> 5;
    const int wm   = wid & 3;
    const int wn   = wid >> 2;
    const int rowBase = blockIdx.y * 128;
    const int colBase = blockIdx.x * 64;

    v8f acc1[2][2], acc2[2][2];
#pragma unroll
    for (int i = 0; i < 2; ++i)
#pragma unroll
        for (int j = 0; j < 2; ++j)
#pragma unroll
            for (int r = 0; r < 8; ++r) { acc1[i][j][r] = 0.f; acc2[i][j][r] = 0.f; }

    const uint4* Ag  = (const uint4*)A;
    const uint4* B1g = (const uint4*)W1t;
    const uint4* B2g = (const uint4*)W2t;

    auto stage = [&](int buf, int k0) {
#pragma unroll
        for (int pck = 0; pck < 2; ++pck) {
            int idx = tid + pck * 256;
            int row = idx >> 2, seg = idx & 3;
            async_b128(lds_off_of(&Alds[buf][row * 5 + seg]),
                       Ag + (((size_t)(rowBase + row) << 10) + k0) / 8 + seg);
        }
        {
            int n = tid >> 2, seg = tid & 3;
            size_t goff = (((size_t)(colBase + n) << 10) + k0) / 8 + seg;
            async_b128(lds_off_of(&B1lds[buf][n * 5 + seg]), B1g + goff);
            async_b128(lds_off_of(&B2lds[buf][n * 5 + seg]), B2g + goff);
        }
    };

    stage(0, 0);
    for (int s = 0; s < 32; ++s) {
        wait_async0();
        __syncthreads();
        if (s + 1 < 32) stage((s + 1) & 1, (s + 1) * 32);

        const uint4* Ab  = Alds[s & 1];
        const uint4* B1b = B1lds[s & 1];
        const uint4* B2b = B2lds[s & 1];
        FragU a[2], b1f[2], b2f[2];
#pragma unroll
        for (int mi = 0; mi < 2; ++mi) {
            int row = wm * 32 + mi * 16 + (lane & 15);
            a[mi].u[0] = Ab[row * 5 + half * 2];
            a[mi].u[1] = Ab[row * 5 + half * 2 + 1];
        }
#pragma unroll
        for (int ni = 0; ni < 2; ++ni) {
            int n = wn * 32 + ni * 16 + (lane & 15);
            b1f[ni].u[0] = B1b[n * 5 + half * 2];
            b1f[ni].u[1] = B1b[n * 5 + half * 2 + 1];
            b2f[ni].u[0] = B2b[n * 5 + half * 2];
            b2f[ni].u[1] = B2b[n * 5 + half * 2 + 1];
        }
#pragma unroll
        for (int mi = 0; mi < 2; ++mi)
#pragma unroll
            for (int ni = 0; ni < 2; ++ni) {
                acc1[mi][ni] = __builtin_amdgcn_wmma_f32_16x16x32_bf16(
                    false, a[mi].v, false, b1f[ni].v, (short)0, acc1[mi][ni], false, false);
                acc2[mi][ni] = __builtin_amdgcn_wmma_f32_16x16x32_bf16(
                    false, a[mi].v, false, b2f[ni].v, (short)0, acc2[mi][ni], false, false);
            }
    }

#pragma unroll
    for (int mi = 0; mi < 2; ++mi) {
#pragma unroll
        for (int ni = 0; ni < 2; ++ni) {
            int col = colBase + wn * 32 + ni * 16 + (lane & 15);
            float bv1 = b1[col], bv2 = b2[col];
#pragma unroll
            for (int r = 0; r < 8; ++r) {
                int row = rowBase + wm * 32 + mi * 16 + half * 8 + r;
                if (row < Mvalid) {
                    C1[(size_t)row * 1024 + col] = f2bf(acc1[mi][ni][r] + bv1);
                    C2[(size_t)row * 1024 + col] = f2bf(acc2[mi][ni][r] + bv2);
                }
            }
        }
    }
}

// ================= LayerNorm over channels (row = one position) =================
__global__ __launch_bounds__(256)
void ln_kernel(const float* __restrict__ X, const float* __restrict__ g,
               const float* __restrict__ bt, uint16_t* __restrict__ Y) {
    __shared__ float red[256];
    const int row = blockIdx.x;
    const float* x = X + (size_t)row * H_;
    float s = 0.f, s2 = 0.f;
    for (int h = threadIdx.x; h < H_; h += 256) { float v = x[h]; s += v; s2 += v * v; }
    red[threadIdx.x] = s; __syncthreads();
    for (int o = 128; o > 0; o >>= 1) { if (threadIdx.x < o) red[threadIdx.x] += red[threadIdx.x + o]; __syncthreads(); }
    float mean = red[0] * (1.0f / H_); __syncthreads();
    red[threadIdx.x] = s2; __syncthreads();
    for (int o = 128; o > 0; o >>= 1) { if (threadIdx.x < o) red[threadIdx.x] += red[threadIdx.x + o]; __syncthreads(); }
    float var = red[0] * (1.0f / H_) - mean * mean;
    float inv = rsqrtf(var + EPS_);
    for (int h = threadIdx.x; h < H_; h += 256) {
        float v = (x[h] - mean) * inv * g[h] + bt[h];
        Y[(size_t)row * H_ + h] = f2bf(v);
    }
}

// ================= ring attention combine (5-way softmax per head) =================
__global__ __launch_bounds__(256)
void ring_combine(const uint16_t* __restrict__ q, const uint16_t* __restrict__ k,
                  const uint16_t* __restrict__ v, const uint16_t* __restrict__ kr,
                  const uint16_t* __restrict__ vr, const float* __restrict__ relay_k,
                  const float* __restrict__ relay_v, uint16_t* __restrict__ attout) {
    const int blk = blockIdx.x;
    const int pos = blk >> 1;
    const int hsel = blk & 1;
    const int b = pos / L_;
    const int l = pos - b * L_;
    const int wid = threadIdx.x >> 5;
    const int lane = threadIdx.x & 31;
    const int nh = hsel * 8 + wid;
    const int h0 = 2 * lane, h1 = 2 * lane + 1;
    const size_t base = (size_t)pos * D_ + nh * HD_;

    uint32_t qp = *(const uint32_t*)(q + base + h0);
    float q0 = bf2f((uint16_t)qp), q1 = bf2f((uint16_t)(qp >> 16));

    float kk0[5], kk1[5], vv0[5], vv1[5];
#pragma unroll
    for (int j = 0; j < 3; ++j) {
        // faithful unfold3 scramble: f = c*3 + u  ->  c' = f % 1024, u' = f / 1024
        {
            int c = nh * HD_ + h0;
            int f = c * 3 + j;
            int cp = f & 1023, up = f >> 10;
            int ls = l + up - 1;
            bool ok = (ls >= 0) && (ls < L_);
            size_t a = (size_t)(b * L_ + (ok ? ls : 0)) * D_ + cp;
            kk0[j] = ok ? bf2f(k[a]) : 0.f;
            vv0[j] = ok ? bf2f(v[a]) : 0.f;
        }
        {
            int c = nh * HD_ + h1;
            int f = c * 3 + j;
            int cp = f & 1023, up = f >> 10;
            int ls = l + up - 1;
            bool ok = (ls >= 0) && (ls < L_);
            size_t a = (size_t)(b * L_ + (ok ? ls : 0)) * D_ + cp;
            kk1[j] = ok ? bf2f(k[a]) : 0.f;
            vv1[j] = ok ? bf2f(v[a]) : 0.f;
        }
    }
    {
        uint32_t kp = *(const uint32_t*)(kr + base + h0);
        uint32_t vp = *(const uint32_t*)(vr + base + h0);
        kk0[3] = bf2f((uint16_t)kp); kk1[3] = bf2f((uint16_t)(kp >> 16));
        vv0[3] = bf2f((uint16_t)vp); vv1[3] = bf2f((uint16_t)(vp >> 16));
        float2 rk = *(const float2*)(relay_k + b * D_ + nh * HD_ + h0);
        float2 rv = *(const float2*)(relay_v + b * D_ + nh * HD_ + h0);
        kk0[4] = rk.x; kk1[4] = rk.y;
        vv0[4] = rv.x; vv1[4] = rv.y;
    }

    float s[5];
#pragma unroll
    for (int j = 0; j < 5; ++j) {
        float pval = q0 * kk0[j] + q1 * kk1[j];
#pragma unroll
        for (int o = 16; o > 0; o >>= 1) pval += __shfl_xor(pval, o);
        s[j] = pval * 0.125f;   // / sqrt(64)
    }
    float m = s[0];
#pragma unroll
    for (int j = 1; j < 5; ++j) m = fmaxf(m, s[j]);
    float e[5], Z = 0.f;
#pragma unroll
    for (int j = 0; j < 5; ++j) { e[j] = __expf(s[j] - m); Z += e[j]; }
    float invZ = 1.f / Z;
    float a0 = 0.f, a1 = 0.f;
#pragma unroll
    for (int j = 0; j < 5; ++j) { a0 += e[j] * vv0[j]; a1 += e[j] * vv1[j]; }
    a0 *= invZ; a1 *= invZ;

    // faithful output scramble: f = nh*(L*HD) + l*HD + h ; d = f/L ; l2 = f%L
    {
        int f = nh * (L_ * HD_) + l * HD_ + h0;
        attout[(size_t)(b * L_ + (f & (L_ - 1))) * D_ + (f >> 11)] = f2bf(a0);
    }
    {
        int f = nh * (L_ * HD_) + l * HD_ + h1;
        attout[(size_t)(b * L_ + (f & (L_ - 1))) * D_ + (f >> 11)] = f2bf(a1);
    }
}

// ================= star attention: masked softmax over 2049 keys + gemv =================
__global__ __launch_bounds__(256)
void star_combine(const float* __restrict__ starq, const uint16_t* __restrict__ sk,
                  const uint16_t* __restrict__ sv, const int* __restrict__ mask,
                  float* __restrict__ att_s) {
    __shared__ float pre[LP_];
    __shared__ float qv[HD_];
    __shared__ float red[256];
    __shared__ float attp[4][HD_];
    const int b = blockIdx.x >> 4;
    const int nh = blockIdx.x & 15;
    const int tid = threadIdx.x;
    if (tid < HD_) qv[tid] = starq[b * D_ + nh * HD_ + tid];
    __syncthreads();

    float lmax = -3.0e38f;
    for (int lp = tid; lp < LP_; lp += 256) {
        float acc = 0.f;
        const uint4* kp = (const uint4*)(sk + (size_t)(b * LP_ + lp) * D_ + nh * HD_);
#pragma unroll
        for (int h8 = 0; h8 < 8; ++h8) {
            V4U t; t.u = kp[h8];
#pragma unroll
            for (int e = 0; e < 8; ++e) acc += qv[h8 * 8 + e] * bf2f(t.h[e]);
        }
        acc *= 0.125f;
        if (lp > 0 && mask[b * L_ + lp - 1]) acc = -3.0e38f;
        pre[lp] = acc;
        lmax = fmaxf(lmax, acc);
    }
    red[tid] = lmax; __syncthreads();
    for (int o = 128; o > 0; o >>= 1) { if (tid < o) red[tid] = fmaxf(red[tid], red[tid + o]); __syncthreads(); }
    float m = red[0]; __syncthreads();
    float lsum = 0.f;
    for (int lp = tid; lp < LP_; lp += 256) { float ev = __expf(pre[lp] - m); pre[lp] = ev; lsum += ev; }
    red[tid] = lsum; __syncthreads();
    for (int o = 128; o > 0; o >>= 1) { if (tid < o) red[tid] += red[tid + o]; __syncthreads(); }
    float Z = red[0];

    const int h = tid & 63, grp = tid >> 6;
    float acc = 0.f;
    for (int lp = grp; lp < LP_; lp += 4)
        acc += pre[lp] * bf2f(sv[(size_t)(b * LP_ + lp) * D_ + nh * HD_ + h]);
    attp[grp][h] = acc; __syncthreads();
    if (grp == 0)
        att_s[b * D_ + nh * HD_ + h] = (attp[0][h] + attp[1][h] + attp[2][h] + attp[3][h]) / Z;
}

// ================= small GEMV: out(B x 1024) = A(B x 1024) * W(1024 x 1024) + bias =================
__global__ __launch_bounds__(256)
void gemv1024(const float* __restrict__ A, const float* __restrict__ W,
              const float* __restrict__ bias, float* __restrict__ out, int do_leaky) {
    const int idx = blockIdx.x * 256 + threadIdx.x;    // B_*1024
    const int b = idx >> 10, n = idx & 1023;
    float acc = 0.f;
    for (int kk = 0; kk < 1024; ++kk) acc += A[b * 1024 + kk] * W[(size_t)kk * 1024 + n];
    acc += bias[n];
    if (do_leaky) acc = leaky(acc);
    out[idx] = acc;
}

// ================= weight transpose + bf16 convert: Wt[n*1024+k] = W[k*1024+n] =================
__global__ __launch_bounds__(256)
void wtconv(const float* __restrict__ W, uint16_t* __restrict__ Wt) {
    const int idx = blockIdx.x * 256 + threadIdx.x;    // 1M
    const int n = idx >> 10, kk = idx & 1023;
    Wt[idx] = f2bf(W[(size_t)kk * 1024 + n]);
}

// ================= misc elementwise kernels =================
__global__ __launch_bounds__(256)
void init_nodes_rembs(const float* __restrict__ data, float* __restrict__ nodes,
                      uint16_t* __restrict__ rembs) {
    size_t idx = (size_t)blockIdx.x * 256 + threadIdx.x;
    float vv = data[idx];
    nodes[idx] = vv;
    rembs[idx] = f2bf(vv);
}

__global__ __launch_bounds__(256)
void relay_init(const float* __restrict__ data, float* __restrict__ relay) {
    const int idx = blockIdx.x * 256 + threadIdx.x;    // B_*H_
    const int b = idx >> 10, h = idx & 1023;
    float s = 0.f;
    for (int l = 0; l < L_; ++l) s += data[(size_t)(b * L_ + l) * H_ + h];
    relay[idx] = s * (1.0f / L_);
}

__global__ __launch_bounds__(256)
void build_y(const float* __restrict__ relay, const float* __restrict__ nodes,
             uint16_t* __restrict__ y) {
    size_t idx = (size_t)blockIdx.x * 256 + threadIdx.x;   // ROWS_Y_PAD*1024
    const int r = (int)(idx >> 10), h = (int)(idx & 1023);
    float vv = 0.f;
    if (r < ROWS_Y) {
        int b = r / LP_, lp = r - b * LP_;
        vv = (lp == 0) ? relay[b * H_ + h] : nodes[(size_t)(b * L_ + lp - 1) * H_ + h];
    }
    y[idx] = f2bf(vv);
}

__global__ __launch_bounds__(256)
void mask_nodes(const int* __restrict__ mask, float* __restrict__ nodes) {
    size_t idx = (size_t)blockIdx.x * 256 + threadIdx.x;
    if (mask[idx >> 10]) nodes[idx] = 0.f;
}

__global__ __launch_bounds__(256)
void final_copy(const float* __restrict__ nodes, const float* __restrict__ relay,
                float* __restrict__ out) {
    size_t idx = (size_t)blockIdx.x * 256 + threadIdx.x;
    const size_t NN = (size_t)ROWS_X * H_;
    if (idx < NN) out[idx] = nodes[idx];
    else if (idx < NN + (size_t)B_ * H_) out[idx] = relay[idx - NN];
}

// ================= host orchestration =================
extern "C" void kernel_launch(void* const* d_in, const int* in_sizes, int n_in,
                              void* d_out, int out_size, void* d_ws, size_t ws_size,
                              hipStream_t stream) {
    (void)in_sizes; (void)n_in; (void)out_size; (void)ws_size;
    const float* data    = (const float*)d_in[0];
    const int*   mask    = (const int*)d_in[1];
    const float* norm_g  = (const float*)d_in[2];
    const float* norm_b  = (const float*)d_in[3];
    const float* ring_wq = (const float*)d_in[4];
    const float* ring_bq = (const float*)d_in[5];
    const float* ring_wk = (const float*)d_in[6];
    const float* ring_bk = (const float*)d_in[7];
    const float* ring_wv = (const float*)d_in[8];
    const float* ring_bv = (const float*)d_in[9];
    const float* star_wq = (const float*)d_in[10];
    const float* star_bq = (const float*)d_in[11];
    const float* star_wk = (const float*)d_in[12];
    const float* star_bk = (const float*)d_in[13];
    const float* star_wv = (const float*)d_in[14];
    const float* star_bv = (const float*)d_in[15];
    const float* ring_wo = (const float*)d_in[16];
    const float* ring_bo = (const float*)d_in[17];
    const float* star_wo = (const float*)d_in[18];
    const float* star_bo = (const float*)d_in[19];

    uint8_t* p = (uint8_t*)d_ws;
    auto take = [&](size_t bytes) -> uint8_t* {
        uint8_t* r = p; p += (bytes + 255) & ~(size_t)255; return r;
    };
    float*    nodes = (float*)   take((size_t)ROWS_X * H_ * 4);
    uint16_t* rembs = (uint16_t*)take((size_t)ROWS_X * H_ * 2);
    uint16_t* xn    = (uint16_t*)take((size_t)ROWS_X * H_ * 2);
    uint16_t* ybuf  = (uint16_t*)take((size_t)ROWS_Y_PAD * H_ * 2);
    uint16_t* qb    = (uint16_t*)take((size_t)ROWS_X * D_ * 2);
    uint16_t* kb    = (uint16_t*)take((size_t)ROWS_X * D_ * 2);
    uint16_t* vb    = (uint16_t*)take((size_t)ROWS_X * D_ * 2);
    uint16_t* krb   = (uint16_t*)take((size_t)ROWS_X * D_ * 2);
    uint16_t* vrb   = (uint16_t*)take((size_t)ROWS_X * D_ * 2);
    uint16_t* attb  = (uint16_t*)take((size_t)ROWS_X * D_ * 2);
    uint16_t* skb   = (uint16_t*)take((size_t)ROWS_Y_PAD * D_ * 2);
    uint16_t* svb   = (uint16_t*)take((size_t)ROWS_Y_PAD * D_ * 2);
    uint16_t* wt    = (uint16_t*)take((size_t)1024 * 1024 * 2);
    uint16_t* wt2   = (uint16_t*)take((size_t)1024 * 1024 * 2);
    float* relay    = (float*)take((size_t)B_ * H_ * 4);
    float* relay_k  = (float*)take((size_t)B_ * D_ * 4);
    float* relay_v  = (float*)take((size_t)B_ * D_ * 4);
    float* starq    = (float*)take((size_t)B_ * D_ * 4);
    float* att_s    = (float*)take((size_t)B_ * D_ * 4);

    const int EW_X  = (ROWS_X * H_) / 256;          // 32768
    const int EW_Y  = (ROWS_Y_PAD * H_) / 256;      // 33280
    const int EW_W  = (1024 * 1024) / 256;          // 4096
    const dim3 gS(8, ROWS_X / 128);                 // single GEMM: 128x128 tiles
    const dim3 gDA(16, ROWS_X / 128);               // dualA: 128x64 tiles
    const dim3 gDB(16, ROWS_Y_PAD / 128);           // dualB (star): 128x64 tiles

    init_nodes_rembs<<<EW_X, 256, 0, stream>>>(data, nodes, rembs);
    relay_init<<<(B_ * H_) / 256, 256, 0, stream>>>(data, relay);

    for (int it = 0; it < ITERS_; ++it) {
        const size_t wOff = (size_t)it * 1024 * 1024;
        const size_t bOff = (size_t)it * 1024;

        // xn = LN(nodes)
        ln_kernel<<<ROWS_X, 256, 0, stream>>>(nodes, norm_g + bOff, norm_b + bOff, xn);

        // q = xn @ Wq + bq
        wtconv<<<EW_W, 256, 0, stream>>>(ring_wq + wOff, wt);
        gemm_bf16<true, false><<<gS, 256, 0, stream>>>(xn, wt, ring_bq + bOff, qb, ROWS_X);
        // k = xn @ Wk + bk ; kr = r_embs @ Wk + bk (shared weight) ; relay_k gemv
        wtconv<<<EW_W, 256, 0, stream>>>(ring_wk + wOff, wt);
        gemm_bf16_dualA<<<gDA, 256, 0, stream>>>(xn, rembs, wt, ring_bk + bOff, kb, krb);
        gemv1024<<<(B_ * D_) / 256, 256, 0, stream>>>(relay, ring_wk + wOff, ring_bk + bOff, relay_k, 0);
        // v / vr / relay_v
        wtconv<<<EW_W, 256, 0, stream>>>(ring_wv + wOff, wt);
        gemm_bf16_dualA<<<gDA, 256, 0, stream>>>(xn, rembs, wt, ring_bv + bOff, vb, vrb);
        gemv1024<<<(B_ * D_) / 256, 256, 0, stream>>>(relay, ring_wv + wOff, ring_bv + bOff, relay_v, 0);

        // ring attention combine -> attb (scrambled position-major)
        ring_combine<<<ROWS_X * 2, 256, 0, stream>>>(qb, kb, vb, krb, vrb, relay_k, relay_v, attb);

        // nodes = leaky(attb @ Wo + bo)
        wtconv<<<EW_W, 256, 0, stream>>>(ring_wo + wOff, wt);
        gemm_bf16<false, true><<<gS, 256, 0, stream>>>(attb, wt, ring_bo + bOff, nodes, ROWS_X);

        // y = [relay ; nodes] per batch (bf16, padded rows zeroed)
        build_y<<<EW_Y, 256, 0, stream>>>(relay, nodes, ybuf);

        // star q (gemv), star k & v in one dual-B GEMM (shared activation tile)
        gemv1024<<<(B_ * D_) / 256, 256, 0, stream>>>(relay, star_wq + wOff, star_bq + bOff, starq, 0);
        wtconv<<<EW_W, 256, 0, stream>>>(star_wk + wOff, wt);
        wtconv<<<EW_W, 256, 0, stream>>>(star_wv + wOff, wt2);
        gemm_bf16_dualB<<<gDB, 256, 0, stream>>>(ybuf, wt, wt2, star_bk + bOff, star_bv + bOff,
                                                 skb, svb, ROWS_Y_PAD);

        // masked softmax attention over 2049 keys
        star_combine<<<B_ * NH_, 256, 0, stream>>>(starq, skb, svb, mask, att_s);

        // relay = leaky(att_s @ star_Wo + star_bo)
        gemv1024<<<(B_ * H_) / 256, 256, 0, stream>>>(att_s, star_wo + wOff, star_bo + bOff, relay, 1);

        // nodes = where(mask, 0, nodes)
        mask_nodes<<<EW_X, 256, 0, stream>>>(mask, nodes);
    }

    const size_t total_out = (size_t)ROWS_X * H_ + (size_t)B_ * H_;
    final_copy<<<(int)((total_out + 255) / 256), 256, 0, stream>>>(nodes, relay, (float*)d_out);
}
